// COCOAloss_55473797595621
// MI455X (gfx1250) — matrix operand order repaired and checked by reading, not compiled, not run on hardware
//
#include <hip/hip_runtime.h>
#include <hip/hip_bf16.h>
#include <cstddef>
#include <cstdint>

// ---------------------------------------------------------------------------
// COCOA loss, MI455X (gfx1250, wave32, WMMA).
//
// B=4096 samples, V=6 views, D=256 dims.  Everything reduces to two scalars:
//   acc[0] = sum_b sum_{v<w} exp(2*(1 - <zn_vb, zn_wb>))          (pos cross)
//   acc[1] = sum_v sum_{b,c} exp(2*<zn_vb, zn_vc>)                (neg full)
// out = S*(6B + 2*acc0)/36 + LAMBDA*(acc1 - 6B*e^2)/(B-1)
// Dominant cost: 6 Gram GEMMs 4096x4096x256 (f16 WMMA) fused with exp-reduce.
// Double-buffered LDS panels staged with CDNA5 async global->LDS copies
// (GLOBAL_LOAD_ASYNC_TO_LDS_B128, ASYNCcnt) overlapping with the WMMA pipe.
// ---------------------------------------------------------------------------

typedef __attribute__((ext_vector_type(16))) _Float16 v16h;
typedef __attribute__((ext_vector_type(8)))  _Float16 v8h;
typedef __attribute__((ext_vector_type(8)))  float    v8f;
typedef int v4i32 __attribute__((vector_size(4 * sizeof(int))));

union F16Frag { v16h v; v8h h[2]; };

#define VIEWS 6
#define DIM   256
#define TK    32      // K step = WMMA K for f16
#define LSTR  40      // LDS row stride in halves (32 + 8 pad -> conflict-free)
#define PANEL (128 * LSTR)

#if defined(__has_builtin)
#  if __has_builtin(__builtin_amdgcn_global_load_async_to_lds_b128)
#    define HAVE_ASYNC_COPY 1
#  endif
#  if __has_builtin(__builtin_amdgcn_s_wait_asynccnt)
#    define HAVE_WAIT_ASYNC 1
#  endif
#endif

__device__ __forceinline__ void copy16_g2l(const void* g, void* l) {
#ifdef HAVE_ASYNC_COPY
    // global -> LDS direct (ASYNCcnt tracked).  AS1 value == generic value;
    // AS3 value == low 32 bits of the generic LDS address (ISA aperture rule).
    __builtin_amdgcn_global_load_async_to_lds_b128(
        (__attribute__((address_space(1))) v4i32*)(uintptr_t)g,
        (__attribute__((address_space(3))) v4i32*)(uint32_t)(uintptr_t)l,
        0, 0);
#else
    *(v8h*)l = *(const v8h*)g;
#endif
}

__device__ __forceinline__ void copy_g2l_wait() {
#ifdef HAVE_ASYNC_COPY
#  ifdef HAVE_WAIT_ASYNC
    __builtin_amdgcn_s_wait_asynccnt(0);
#  else
    asm volatile("s_wait_asynccnt 0" ::: "memory");
#  endif
#endif
}

// --------------------------- accumulator init ------------------------------
__global__ void zero_acc_kernel(float* acc) {
    if (threadIdx.x < 2) acc[threadIdx.x] = 0.0f;
}

// --------------------- normalize + f16 convert + pos -----------------------
// One block per sample b; 6 waves (192 threads), wave w handles view w.
__global__ __launch_bounds__(192)
void norm_pos_kernel(const float* __restrict__ z, _Float16* __restrict__ zn,
                     float* __restrict__ acc, int Bsz) {
    __shared__ __align__(16) float sh[VIEWS][DIM];

    const int b    = blockIdx.x;
    const int wave = threadIdx.x >> 5;   // view index
    const int lane = threadIdx.x & 31;
    const int d0   = lane * 8;           // 32 lanes * 8 = 256 dims

    const float* src = z + ((size_t)b * VIEWS + wave) * DIM + d0;
    float x[8];
    {
        float4 p0 = *(const float4*)(src + 0);
        float4 p1 = *(const float4*)(src + 4);
        x[0]=p0.x; x[1]=p0.y; x[2]=p0.z; x[3]=p0.w;
        x[4]=p1.x; x[5]=p1.y; x[6]=p1.z; x[7]=p1.w;
    }
    float ssq = 0.0f;
#pragma unroll
    for (int k = 0; k < 8; ++k) ssq += x[k] * x[k];
#pragma unroll
    for (int off = 16; off; off >>= 1) ssq += __shfl_xor(ssq, off, 32);
    const float rn = rsqrtf(ssq);

    v8h hv;
#pragma unroll
    for (int k = 0; k < 8; ++k) {
        float xn = x[k] * rn;
        sh[wave][d0 + k] = xn;
        hv[k] = (_Float16)xn;
    }
    // zn layout: [view][b][d], row-major, f16 (WMMA feed)
    *(v8h*)(zn + ((size_t)wave * Bsz + b) * DIM + d0) = hv;

    __syncthreads();

    // wave 0: 15 unique cross-view dot products -> pos cross terms
    if (wave == 0) {
        float se = 0.0f;
#pragma unroll
        for (int v = 0; v < VIEWS; ++v) {
#pragma unroll
            for (int w = v + 1; w < VIEWS; ++w) {
                float p = 0.0f;
#pragma unroll
                for (int k = 0; k < 8; ++k)
                    p += sh[v][d0 + k] * sh[w][d0 + k];
#pragma unroll
                for (int off = 16; off; off >>= 1) p += __shfl_xor(p, off, 32);
                se += __expf(2.0f * (1.0f - p));   // (1-sim)/T, T=0.5
            }
        }
        if (lane == 0) unsafeAtomicAdd(&acc[0], se);
    }
}

// --------------- fused Gram GEMM + exp + global reduction ------------------
// grid = (B/128, B/128, V); block = 256 threads = 8 waves (4 M-slices x 2 N).
// Each wave: 32x64 of C via 2x4 f16 WMMA accumulators; K loop over 256 with
// double-buffered async-staged 128x32 LDS panels.
__global__ __launch_bounds__(256)
void gram_exp_kernel(const _Float16* __restrict__ zn, float* __restrict__ acc,
                     int Bsz) {
    __shared__ __align__(16) _Float16 Ap[2][PANEL];
    __shared__ __align__(16) _Float16 Bp[2][PANEL];
    __shared__ float red[8];

    const int v     = blockIdx.z;
    const int mbase = blockIdx.x * 128;
    const int nbase = blockIdx.y * 128;
    const _Float16* base = zn + (size_t)v * Bsz * DIM;

    const int tid  = threadIdx.x;
    const int lane = tid & 31;
    const int wave = tid >> 5;
    const int wm   = wave & 3;      // M slice: wm*32
    const int wn   = wave >> 2;     // N slice: wn*64
    const int lrow = lane & 15;
    const bool hi  = lane >= 16;

    // per-thread staging coordinates: 512 v8h chunks per panel / 256 threads
    const int sr0 = tid >> 2;               // rows tid/4 and tid/4 + 64
    const int sc  = (tid & 3) * 8;          // k-chunk within row

    v8f c[2][4] = {};

    // prologue: stage k0 = 0 into buffer 0
    {
        copy16_g2l(&base[(size_t)(mbase + sr0) * DIM + sc],      &Ap[0][sr0 * LSTR + sc]);
        copy16_g2l(&base[(size_t)(mbase + sr0 + 64) * DIM + sc], &Ap[0][(sr0 + 64) * LSTR + sc]);
        copy16_g2l(&base[(size_t)(nbase + sr0) * DIM + sc],      &Bp[0][sr0 * LSTR + sc]);
        copy16_g2l(&base[(size_t)(nbase + sr0 + 64) * DIM + sc], &Bp[0][(sr0 + 64) * LSTR + sc]);
    }

#pragma unroll
    for (int step = 0; step < DIM / TK; ++step) {
        const int cur = step & 1;
        const int nxt = cur ^ 1;

        copy_g2l_wait();       // this wave's async writes for buf[cur] done
        __syncthreads();       // all waves' staging of buf[cur] visible

        if (step + 1 < DIM / TK) {
            const int k1 = (step + 1) * TK;
            copy16_g2l(&base[(size_t)(mbase + sr0) * DIM + k1 + sc],      &Ap[nxt][sr0 * LSTR + sc]);
            copy16_g2l(&base[(size_t)(mbase + sr0 + 64) * DIM + k1 + sc], &Ap[nxt][(sr0 + 64) * LSTR + sc]);
            copy16_g2l(&base[(size_t)(nbase + sr0) * DIM + k1 + sc],      &Bp[nxt][sr0 * LSTR + sc]);
            copy16_g2l(&base[(size_t)(nbase + sr0 + 64) * DIM + k1 + sc], &Bp[nxt][(sr0 + 64) * LSTR + sc]);
        }

        // A fragments (ISA 7.12.2, 16-bit A 16x32): lane<16 -> K {0..7,16..23},
        // lane>=16 -> K {8..15,24..31}; each group is one ds_load_b128.
        F16Frag a[2];
#pragma unroll
        for (int i = 0; i < 2; ++i) {
            const int row = wm * 32 + i * 16 + lrow;
            a[i].h[0] = *(const v8h*)&Ap[cur][row * LSTR + (hi ? 8 : 0)];
            a[i].h[1] = *(const v8h*)&Ap[cur][row * LSTR + (hi ? 24 : 16)];
        }
        // B fragments (same K order as A):
        // lane<16 -> K {0..7,16..23}, lane>=16 -> K {8..15,24..31}.
        F16Frag bf[4];
#pragma unroll
        for (int j = 0; j < 4; ++j) {
            const int col = wn * 64 + j * 16 + lrow;
            bf[j].h[0] = *(const v8h*)&Bp[cur][col * LSTR + (hi ? 8 : 0)];
            bf[j].h[1] = *(const v8h*)&Bp[cur][col * LSTR + (hi ? 24 : 16)];
        }
#pragma unroll
        for (int i = 0; i < 2; ++i)
#pragma unroll
            for (int j = 0; j < 4; ++j)
                c[i][j] = __builtin_amdgcn_wmma_f32_16x16x32_f16(
                    false, a[i].v, false, bf[j].v,
                    (short)0, c[i][j], false, false);

        __syncthreads();       // compute done before next staging overwrites
    }

    // epilogue: sum exp(sim / T) over this block's 128x128 tile (T = 0.5)
    float s = 0.0f;
#pragma unroll
    for (int i = 0; i < 2; ++i)
#pragma unroll
        for (int j = 0; j < 4; ++j)
#pragma unroll
            for (int e = 0; e < 8; ++e)
                s += __expf(2.0f * c[i][j][e]);

#pragma unroll
    for (int off = 16; off; off >>= 1) s += __shfl_xor(s, off, 32);
    if (lane == 0) red[wave] = s;
    __syncthreads();
    if (tid == 0) {
        float t = 0.0f;
#pragma unroll
        for (int w = 0; w < 8; ++w) t += red[w];
        unsafeAtomicAdd(&acc[1], t);   // global_atomic_add_f32
    }
}

// ------------------------------ finalize -----------------------------------
__global__ void finalize_kernel(const float* __restrict__ acc,
                                float* __restrict__ out, int Bsz) {
    const float e2  = __expf(2.0f);                       // diagonal of neg
    const float pos = (6.0f * (float)Bsz + 2.0f * acc[0]) * (1.0f / 36.0f);
    const float neg = (acc[1] - 6.0f * (float)Bsz * e2) / (float)(Bsz - 1);
    out[0] = pos * (1.0f / 32.0f) + 0.0039f * neg;
}

// ------------------------------- launcher ----------------------------------
extern "C" void kernel_launch(void* const* d_in, const int* in_sizes, int n_in,
                              void* d_out, int out_size, void* d_ws, size_t ws_size,
                              hipStream_t stream) {
    const float* z   = (const float*)d_in[0];
    float*       out = (float*)d_out;
    const int Bsz = in_sizes[0] / (VIEWS * DIM);   // 4096

    float*     acc = (float*)d_ws;                          // 2 f32 accumulators
    _Float16*  zn  = (_Float16*)((char*)d_ws + 256);        // [V][B][D] f16

    zero_acc_kernel<<<1, 32, 0, stream>>>(acc);
    norm_pos_kernel<<<Bsz, 192, 0, stream>>>(z, zn, acc, Bsz);
    dim3 grid(Bsz / 128, Bsz / 128, VIEWS);
    gram_exp_kernel<<<grid, 256, 0, stream>>>(zn, acc, Bsz);
    finalize_kernel<<<1, 1, 0, stream>>>(acc, out, Bsz);
}
